// Informer_Track_73796128080011
// MI455X (gfx1250) — matrix operand order repaired and checked by reading, not compiled, not run on hardware
//
#include <hip/hip_runtime.h>
#include <hip/hip_bf16.h>
#include <math.h>

// ---------------------------------------------------------------------------
// Informer forward pass for gfx1250 (MI455X).
// All dense GEMMs run through a single wave32 WMMA kernel:
//   v_wmma_f32_16x16x32_f16, 8 waves/block, 64x64 block tile,
//   double-buffered LDS software pipeline (1 barrier per K-step),
//   branch-free vectorized staging; async global->LDS (ASYNCcnt) staging.
// K is always a multiple of 32; M/N edges via clamping + masked epilogue.
// Requires ws_size >= ~403 MiB.
// ---------------------------------------------------------------------------

typedef _Float16 h16;
typedef __attribute__((ext_vector_type(16))) _Float16 v16h;
typedef __attribute__((ext_vector_type(8)))  _Float16 v8h;
typedef __attribute__((ext_vector_type(8)))  float    v8f;
typedef __attribute__((ext_vector_type(4)))  int      int4v;

#if defined(__AMDGCN__) && __has_builtin(__builtin_amdgcn_global_load_async_to_lds_b128) && __has_builtin(__builtin_amdgcn_s_wait_asynccnt)
#define ASYNC_LDS 1
typedef __attribute__((address_space(1))) int4v g4i;   // global int4
typedef __attribute__((address_space(3))) int4v l4i;   // LDS int4
#else
#define ASYNC_LDS 0
#endif

#define NB   4
#define NH   8
#define HD   64
#define DM   512
#define DFF  2048
#define SEQ  2048
#define DECL 1536

// ======================= generic batched WMMA GEMM =========================
// C[m,n] = act( alpha * sum_k A[m,k]*B[k,n] + bias[n] ) + Cold[m,n]
// A element (m,k) at  A + a0 + zb*aOb + zh*aOh + m*aM + k        (f16, k-contig)
// B element (k,n) at  B + b0 + zb*bOb + zh*bOh + n*bN + k*bK     (f16)
//   requires bN==1 (n-contiguous) or bK==1 (k-contiguous)
// C element (m,n) at  C + c0 + zb*cOb + zh*cOh + m*cM + n        (f32, opt f16)
// K must be a multiple of 32. All strides are multiples of 8 halves.
struct GP {
  const h16* A; const h16* B; float* C; h16* C16;
  const float* bias; const float* Cold;
  long a0, b0, c0;
  long aOb, aOh, bOb, bOh, cOb, cOh;
  int aM, bN, bK, cM;
  int M, N, K, Hdiv, act;   // act: 0=none, 1=gelu(exact)
  float alpha;
};

__global__ __launch_bounds__(256) void k_gemm(GP p) {
  __shared__ __align__(16) h16 sA[2][64][40];   // double-buffered 64m x 32k
  __shared__ __align__(16) h16 sB[2][64][40];   // double-buffered 64n x 32k
  const int tid = threadIdx.x;
  const int z  = blockIdx.z;
  const int zb = z / p.Hdiv, zh = z % p.Hdiv;
  const long aBase = p.a0 + (long)zb * p.aOb + (long)zh * p.aOh;
  const long bBase = p.b0 + (long)zb * p.bOb + (long)zh * p.bOh;
  const long cBase = p.c0 + (long)zb * p.cOb + (long)zh * p.cOh;
  const int m0 = blockIdx.x * 64;
  const int n0 = blockIdx.y * 64;
  const int wave = tid >> 5, lane = tid & 31;
  const int wm = wave >> 1, wn = wave & 1;
  const int lm = lane & 15, lh = lane >> 4;
  v8f acc0 = {0.f,0.f,0.f,0.f,0.f,0.f,0.f,0.f};
  v8f acc1 = acc0;

  // staging assignments (branch-free; rows clamped into valid range)
  const int arow = tid >> 2, aseg = (tid & 3) * 8;            // A: 64r x 4seg
  int agm = m0 + arow; if (agm > p.M - 1) agm = p.M - 1;
  const h16* aptr = p.A + aBase + (long)agm * p.aM + aseg;
  const bool bkc = (p.bK == 1);
  // path1 (k-contig): 64 n-rows x 4 segs of 8 k
  int b1n = n0 + (tid >> 2); if (b1n > p.N - 1) b1n = p.N - 1;
  const int b1seg = (tid & 3) * 8;
  const h16* b1ptr = p.B + bBase + (long)b1n * p.bN + b1seg;
  // path2 (n-contig): 32 k-rows x 8 segs of 8 n  (slop-padded weight buffers)
  const int b2k = tid >> 3, b2ns = (tid & 7) * 8;
  const h16* b2ptr = p.B + bBase + (long)b2k * p.bK + (n0 + b2ns);

  // fragment build + 2 WMMAs from LDS buffer `cur`
  auto compute = [&](int cur) {
    const h16* ar  = &sA[cur][wm * 16 + lm][0];
    const h16* br0 = &sB[cur][wn * 32 + lm][0];
    const h16* br1 = &sB[cur][wn * 32 + 16 + lm][0];
    v8h alo = *(const v8h*)(ar + 16 * lh);       // A: K = 16h+e
    v8h ahi = *(const v8h*)(ar + 16 * lh + 8);
    v8h b0l = *(const v8h*)(br0 + 16 * lh);      // B: K = 16h+e, N = lane%16
    v8h b0h = *(const v8h*)(br0 + 16 * lh + 8);
    v8h b1l = *(const v8h*)(br1 + 16 * lh);
    v8h b1h = *(const v8h*)(br1 + 16 * lh + 8);
    v16h a, b0, b1;
    #pragma unroll
    for (int i2 = 0; i2 < 8; ++i2) {
      a[i2] = alo[i2];  a[8 + i2] = ahi[i2];
      b0[i2] = b0l[i2]; b0[8 + i2] = b0h[i2];
      b1[i2] = b1l[i2]; b1[8 + i2] = b1h[i2];
    }
    acc0 = __builtin_amdgcn_wmma_f32_16x16x32_f16(false, a, false, b0, (short)0, acc0, false, false);
    acc1 = __builtin_amdgcn_wmma_f32_16x16x32_f16(false, a, false, b1, (short)0, acc1, false, false);
  };

  const int nk = p.K >> 5;
  { // prologue: stage tile 0 into buffer 0
#if ASYNC_LDS
    __builtin_amdgcn_global_load_async_to_lds_b128(
        (g4i*)(void*)aptr, (l4i*)(void*)&sA[0][arow][aseg], 0, 0);
#else
    *(v8h*)&sA[0][arow][aseg] = *(const v8h*)aptr;
#endif
    if (bkc) {
#if ASYNC_LDS
      __builtin_amdgcn_global_load_async_to_lds_b128(
          (g4i*)(void*)b1ptr, (l4i*)(void*)&sB[0][tid >> 2][b1seg], 0, 0);
#else
      *(v8h*)&sB[0][tid >> 2][b1seg] = *(const v8h*)b1ptr;
#endif
    } else {
      v8h bv = *(const v8h*)b2ptr;
      #pragma unroll
      for (int i2 = 0; i2 < 8; ++i2) sB[0][b2ns + i2][b2k] = bv[i2];
    }
#if ASYNC_LDS
    __builtin_amdgcn_s_wait_asynccnt(0);
#endif
    __syncthreads();
  }
  for (int it = 1; it < nk; ++it) {
    const int cur = (it - 1) & 1, nxt = it & 1;
    const int kc = it << 5;
    // issue next-tile global loads first so they overlap the WMMAs
    __builtin_prefetch(aptr + kc + 32, 0, 1);    // -> global_prefetch_b8
#if ASYNC_LDS
    __builtin_amdgcn_global_load_async_to_lds_b128(
        (g4i*)(void*)(aptr + kc), (l4i*)(void*)&sA[nxt][arow][aseg], 0, 0);
    if (bkc)
      __builtin_amdgcn_global_load_async_to_lds_b128(
          (g4i*)(void*)(b1ptr + kc), (l4i*)(void*)&sB[nxt][tid >> 2][b1seg], 0, 0);
#else
    v8h areg = *(const v8h*)(aptr + kc);
    v8h b1reg = {};
    if (bkc) b1reg = *(const v8h*)(b1ptr + kc);
#endif
    v8h b2reg = {};
    if (!bkc) b2reg = *(const v8h*)(b2ptr + (long)kc * p.bK);
    // compute previous tile while loads are in flight
    compute(cur);
    // commit staged data to LDS buffer `nxt`
#if !ASYNC_LDS
    *(v8h*)&sA[nxt][arow][aseg] = areg;
    if (bkc) *(v8h*)&sB[nxt][tid >> 2][b1seg] = b1reg;
#endif
    if (!bkc) {
      #pragma unroll
      for (int i2 = 0; i2 < 8; ++i2) sB[nxt][b2ns + i2][b2k] = b2reg[i2];
    }
#if ASYNC_LDS
    __builtin_amdgcn_s_wait_asynccnt(0);
#endif
    __syncthreads();
  }
  compute((nk - 1) & 1);   // tail tile

  // epilogue: D rows M = j + 8h, cols N = lane%16 (two 16-col groups)
  #pragma unroll
  for (int j = 0; j < 8; ++j) {
    int gm = m0 + wm * 16 + j + 8 * lh;
    if (gm >= p.M) continue;
    long rbase = cBase + (long)gm * p.cM;
    #pragma unroll
    for (int t2 = 0; t2 < 2; ++t2) {
      int gn = n0 + wn * 32 + t2 * 16 + lm;
      if (gn >= p.N) continue;
      float v = (t2 ? acc1[j] : acc0[j]) * p.alpha;
      if (p.bias) v += p.bias[gn];
      if (p.act == 1) v = 0.5f * v * (1.0f + erff(v * 0.70710678f));
      long co = rbase + gn;
      if (p.Cold) v += p.Cold[co];
      p.C[co] = v;
      if (p.C16) p.C16[co] = (h16)v;
    }
  }
}

// ======================= pointwise / reduce kernels ========================
__global__ void k_cast(const float* __restrict__ s, h16* __restrict__ d, long n) {
  long i = (long)blockIdx.x * 256 + threadIdx.x;
  if (i < n) d[i] = (h16)s[i];
}

// circular pad along length with channel zero-padding Cin -> Cpad
__global__ void k_pad_circ(const float* __restrict__ x, h16* __restrict__ xp,
                           int L, int Cin, int Cpad, long total) {
  long i = (long)blockIdx.x * 256 + threadIdx.x;
  if (i >= total) return;
  int c = (int)(i % Cpad); long t = i / Cpad;
  int lp = (int)(t % (L + 2)); int b = (int)(t / (L + 2));
  int l = (lp == 0) ? (L - 1) : ((lp == L + 1) ? 0 : (lp - 1));
  xp[i] = (c < Cin) ? (h16)x[((long)b * L + l) * Cin + c] : (h16)0.f;
}

// conv weight W[o][c][k] (f32) -> out[k][c][o] f16, c zero-padded to Cpad
__global__ void k_repack_convw(const float* __restrict__ W, h16* __restrict__ out,
                               int Cin, int Cpad, long total) {
  long i = (long)blockIdx.x * 256 + threadIdx.x;
  if (i >= total) return;
  int o = (int)(i % DM); long t = i / DM;
  int c = (int)(t % Cpad); int k = (int)(t / Cpad);
  out[i] = (c < Cin) ? (h16)W[((long)o * Cin + c) * 3 + k] : (h16)0.f;
}

// W[K][N] f32 -> out[K][Npad] f16 zero-padded columns
__global__ void k_pad_w(const float* __restrict__ W, h16* __restrict__ out,
                        int N, int Npad, long total) {
  long i = (long)blockIdx.x * 256 + threadIdx.x;
  if (i >= total) return;
  int n = (int)(i % Npad); long k = i / Npad;
  out[i] = (n < N) ? (h16)W[k * N + n] : (h16)0.f;
}

// add positional + temporal sinusoid embeddings in-place (f32) + f16 mirror
__global__ void k_embed_add(float* __restrict__ y, h16* __restrict__ y16,
                            const int* __restrict__ mark, int L, long total) {
  long i = (long)blockIdx.x * 256 + threadIdx.x;
  if (i >= total) return;
  int c = (int)(i & 511); long t = i >> 9;
  int l = (int)(t % L); int b = (int)(t / L);
  float dv = __expf(-(float)(c & ~1) * (9.210340372f / 512.0f)); // ln(1e4)/d
  const int* mk = mark + ((long)b * L + l) * 4;
  float p0 = (float)l, p1 = (float)mk[3], p2 = (float)mk[2], p3 = (float)mk[1], p4 = (float)mk[0];
  float s;
  if (c & 1) s = cosf(p0*dv)+cosf(p1*dv)+cosf(p2*dv)+cosf(p3*dv)+cosf(p4*dv);
  else       s = sinf(p0*dv)+sinf(p1*dv)+sinf(p2*dv)+sinf(p3*dv)+sinf(p4*dv);
  float v = y[i] + s;
  y[i] = v; y16[i] = (h16)v;
}

// out = LayerNorm(x + a) * g + b  over last dim (512); optional f16 mirror
__global__ __launch_bounds__(256) void k_add_ln(const float* __restrict__ x, const float* __restrict__ a,
                        const float* __restrict__ g, const float* __restrict__ b,
                        float* __restrict__ o, h16* __restrict__ o16) {
  int r = blockIdx.x, t = threadIdx.x;
  long base = (long)r * 512;
  float v0 = x[base + t]       + (a ? a[base + t]       : 0.f);
  float v1 = x[base + t + 256] + (a ? a[base + t + 256] : 0.f);
  __shared__ float s1[256], s2[256];
  s1[t] = v0 + v1; s2[t] = v0 * v0 + v1 * v1;
  __syncthreads();
  for (int st = 128; st > 0; st >>= 1) { if (t < st) { s1[t] += s1[t+st]; s2[t] += s2[t+st]; } __syncthreads(); }
  float m  = s1[0] * (1.f / 512.f);
  float vr = s2[0] * (1.f / 512.f) - m * m;
  float rs = rsqrtf(vr + 1e-5f);
  float w0 = (v0 - m) * rs * g[t]       + b[t];
  float w1 = (v1 - m) * rs * g[t + 256] + b[t + 256];
  o[base + t] = w0; o[base + t + 256] = w1;
  if (o16) { o16[base + t] = (h16)w0; o16[base + t + 256] = (h16)w1; }
}

// row softmax; cols masked beyond per-row limit (inclusive). probs -> f16
__global__ __launch_bounds__(256) void k_softmax(const float* __restrict__ sc, h16* __restrict__ pr,
                         const int* __restrict__ limit, int cols) {
  long r = blockIdx.x; int t = threadIdx.x;
  int lim = limit ? limit[r] : (cols - 1);
  const float* row = sc + r * (long)cols;
  __shared__ float red[256];
  float mx = -3.0e38f;
  for (int c = t; c <= lim; c += 256) mx = fmaxf(mx, row[c]);
  red[t] = mx; __syncthreads();
  for (int st = 128; st > 0; st >>= 1) { if (t < st) red[t] = fmaxf(red[t], red[t+st]); __syncthreads(); }
  mx = red[0]; __syncthreads();
  float sum = 0.f;
  for (int c = t; c <= lim; c += 256) sum += __expf(row[c] - mx);
  red[t] = sum; __syncthreads();
  for (int st = 128; st > 0; st >>= 1) { if (t < st) red[t] += red[t+st]; __syncthreads(); }
  float inv = 1.0f / red[0];
  h16* prow = pr + r * (long)cols;
  for (int c = t; c < cols; c += 256)
    prow[c] = (h16)((c <= lim) ? __expf(row[c] - mx) * inv : 0.f);
}

// ProbSparse measure M = max_s(q.k_s) - sum_s(q.k_s)/Lk over U hashed samples
__global__ void k_sample_M(const h16* __restrict__ Q, const h16* __restrict__ K,
                           float* __restrict__ M, int Lq, int Lk, int U, int H) {
  int l = blockIdx.x, z = blockIdx.y, t = threadIdx.x;
  int b = z / H, h = z % H;
  const h16* q = Q + (((long)b * Lq + l) * H + h) * HD;
  float dot = -3.0e38f, val = 0.f;
  if (t < U) {
    unsigned idx = ((unsigned)l * 1103515245u + (unsigned)t * 2654435761u + 12345u) % (unsigned)Lk;
    const h16* kk = K + (((long)b * Lk + (int)idx) * H + h) * HD;
    float s = 0.f;
    for (int d = 0; d < HD; ++d) s += (float)q[d] * (float)kk[d];
    dot = s; val = s;
  }
  __shared__ float smax[64], ssum[64];
  smax[t] = dot; ssum[t] = val; __syncthreads();
  for (int st = 32; st > 0; st >>= 1) { if (t < st) { smax[t] = fmaxf(smax[t], smax[t+st]); ssum[t] += ssum[t+st]; } __syncthreads(); }
  if (t == 0) M[(long)z * Lq + l] = smax[0] - ssum[0] / (float)Lk;
}

// iterative top-u argmax selection (destroys M)
__global__ __launch_bounds__(256) void k_topk(float* __restrict__ M, int* __restrict__ top, int Lq, int u) {
  int z = blockIdx.x, t = threadIdx.x;
  float* m = M + (long)z * Lq;
  __shared__ float bv[256]; __shared__ int bi[256];
  for (int it = 0; it < u; ++it) {
    float best = -3.0e38f; int bidx = 0;
    for (int c = t; c < Lq; c += 256) { float v = m[c]; if (v > best) { best = v; bidx = c; } }
    bv[t] = best; bi[t] = bidx; __syncthreads();
    for (int st = 128; st > 0; st >>= 1) { if (t < st && bv[t+st] > bv[t]) { bv[t] = bv[t+st]; bi[t] = bi[t+st]; } __syncthreads(); }
    if (t == 0) { top[(long)z * u + it] = bi[0]; m[bi[0]] = -3.0e38f; }
    __syncthreads();
  }
}

__global__ void k_gather_q(const h16* __restrict__ Q, const int* __restrict__ top,
                           h16* __restrict__ qr, int L, int H, int u, long total) {
  long i = (long)blockIdx.x * 256 + threadIdx.x;
  if (i >= total) return;
  int d = (int)(i & 63); long t = i >> 6;
  int ui = (int)(t % u); int z = (int)(t / u);
  int b = z / H, h = z % H;
  int l = top[(long)z * u + ui];
  qr[i] = Q[(((long)b * L + l) * H + h) * HD + d];
}

__global__ void k_mean_v(const float* __restrict__ V, float* __restrict__ mv, int L, int H) {
  int z = blockIdx.x, t = threadIdx.x;   // 64 threads
  int b = z / H, h = z % H;
  float s = 0.f;
  for (int l = 0; l < L; ++l) s += V[((long)b * L + l) * DM + h * HD + t];
  mv[(long)z * HD + t] = s / (float)L;
}

__global__ void k_fill_ctx(float* __restrict__ ctx, const float* __restrict__ mv,
                           int L, long total) {
  long i = (long)blockIdx.x * 256 + threadIdx.x;
  if (i >= total) return;
  int d = (int)(i & 63);
  long z = i / ((long)L * HD);
  ctx[i] = mv[z * HD + d];
}

__global__ void k_cumsum_ctx(const float* __restrict__ V, float* __restrict__ ctx, int L, int H) {
  int z = blockIdx.x, t = threadIdx.x;   // 64 threads
  int b = z / H, h = z % H;
  float acc = 0.f;
  for (int l = 0; l < L; ++l) {
    acc += V[((long)b * L + l) * DM + h * HD + t];
    ctx[((long)z * L + l) * HD + t] = acc;
  }
}

__global__ void k_scatter_ctx(float* __restrict__ ctx, const float* __restrict__ upd,
                              const int* __restrict__ top, int L, int u, long total) {
  long i = (long)blockIdx.x * 256 + threadIdx.x;
  if (i >= total) return;
  int d = (int)(i & 63); long t = i >> 6;
  int ui = (int)(t % u); int z = (int)(t / u);
  int l = top[(long)z * u + ui];
  ctx[((long)z * L + l) * HD + d] = upd[i];
}

// ctx (B,H,L,D) -> rows f16 (B*L, 512).  mix=1: flat reshape, mix=0: permute
__global__ void k_rows_from_ctx(const float* __restrict__ ctx, h16* __restrict__ rows,
                                int L, int H, int mix, long total) {
  long i = (long)blockIdx.x * 256 + threadIdx.x;
  if (i >= total) return;
  if (mix) { rows[i] = (h16)ctx[i]; return; }
  int cc = (int)(i & 511); long t = i >> 9;
  int l = (int)(t % L); int b = (int)(t / L);
  int h = cc >> 6, d = cc & 63;
  rows[i] = (h16)ctx[(((long)b * H + h) * L + l) * HD + d];
}

__global__ __launch_bounds__(256) void k_bn_stats(const float* __restrict__ y, float* __restrict__ mean,
                          float* __restrict__ var, long rows) {
  int c = blockIdx.x, t = threadIdx.x;
  float s = 0.f, s2 = 0.f;
  for (long r = t; r < rows; r += 256) { float v = y[r * DM + c]; s += v; s2 += v * v; }
  __shared__ float r1[256], r2[256];
  r1[t] = s; r2[t] = s2; __syncthreads();
  for (int st = 128; st > 0; st >>= 1) { if (t < st) { r1[t] += r1[t+st]; r2[t] += r2[t+st]; } __syncthreads(); }
  if (t == 0) { float m = r1[0] / (float)rows; mean[c] = m; var[c] = r2[0] / (float)rows - m * m; }
}

__global__ void k_bn_elu(float* __restrict__ y, const float* __restrict__ m, const float* __restrict__ v,
                         const float* __restrict__ g, const float* __restrict__ b, long total) {
  long i = (long)blockIdx.x * 256 + threadIdx.x;
  if (i >= total) return;
  int c = (int)(i & 511);
  float x = (y[i] - m[c]) * rsqrtf(v[c] + 1e-5f) * g[c] + b[c];
  y[i] = (x > 0.f) ? x : expm1f(x);
}

// maxpool window 3 stride 2 pad 1 along length; writes f32 + f16
__global__ void k_maxpool(const float* __restrict__ y, float* __restrict__ o, h16* __restrict__ o16,
                          int L, long total) {
  long i = (long)blockIdx.x * 256 + threadIdx.x;
  if (i >= total) return;
  int Lout = L / 2;
  int c = (int)(i & 511); long t = i >> 9;
  int lo = (int)(t % Lout); int b = (int)(t / Lout);
  float best = -3.0e38f;
  for (int dl = -1; dl <= 1; ++dl) {
    int l = 2 * lo + dl;
    if (l >= 0 && l < L) best = fmaxf(best, y[((long)b * L + l) * DM + c]);
  }
  o[i] = best; o16[i] = (h16)best;
}

// ============================ host orchestration ===========================
static inline dim3 g1(long total) { return dim3((unsigned)((total + 255) / 256)); }

static void gemm_launch(hipStream_t s, const GP& p, int batch) {
  dim3 grid((p.M + 63) / 64, (p.N + 63) / 64, batch);
  k_gemm<<<grid, 256, 0, s>>>(p);
}

// A[M,K] (contig f16) @ W[K,N] (f16 row-major, n-contiguous) + bias
static GP gp_linear(const h16* A, const h16* Wt, const float* bias,
                    float* C, h16* C16, int M, int N, int K, int act) {
  GP p{}; p.A = A; p.aM = K;
  p.B = Wt; p.bN = 1; p.bK = N;
  p.C = C; p.C16 = C16; p.cM = N;
  p.bias = bias; p.M = M; p.N = N; p.K = K; p.Hdiv = 1; p.act = act; p.alpha = 1.f;
  return p;
}

struct AttnW { const h16 *wq, *wk, *wv, *wo; const float *bq, *bk, *bv, *bo; };

// -------- param leaf indices (jax pytree: dict keys sorted alphabetically) --
enum {
  I_DEC_EMB_B = 0, I_DEC_EMB_W = 1,
  I_DEC_L0 = 2, I_DEC_L1 = 28,     // per-layer stride 26
  I_DEC_NORM_B = 54, I_DEC_NORM_G = 55,
  I_ENC_CONV0 = 56, I_ENC_CONV1 = 60,   // b, bn_b, bn_g, w
  I_ENC_EMB_B = 64, I_ENC_EMB_W = 65,
  I_ENC_L0 = 66, I_ENC_L1 = 82, I_ENC_L2 = 98,  // per-layer stride 16
  I_ENC_NORM_B = 114, I_ENC_NORM_G = 115,
  I_PROJ_B = 116, I_PROJ_W = 117,
  I_XENC = 118, I_XMARK_ENC = 119, I_XDEC = 120, I_XMARK_DEC = 121
};

extern "C" void kernel_launch(void* const* d_in, const int* in_sizes, int n_in,
                              void* d_out, int out_size, void* d_ws, size_t ws_size,
                              hipStream_t stream) {
  (void)in_sizes; (void)n_in; (void)out_size; (void)ws_size;
  auto F = [&](int i) -> const float* { return (const float*)d_in[i]; };
  auto Ii = [&](int i) -> const int*  { return (const int*)d_in[i]; };

  // ---- scratch bump allocator ----
  char* wsb = (char*)d_ws;
  size_t off = 0;
  auto balloc = [&](size_t bytes) -> void* {
    void* p = wsb + off; off += (bytes + 255) & ~(size_t)255; return p;
  };
  const size_t MBy = 1ull << 20;
  h16*   wbuf   = (h16*)  balloc(44 * MBy);                 // f16 weights (+slop)
  float* xe32   = (float*)balloc((size_t)NB*SEQ*DM*4);      // main activation
  h16*   xe16   = (h16*)  balloc((size_t)NB*SEQ*DM*2);
  float* xt32   = (float*)balloc((size_t)NB*SEQ*DM*4);
  h16*   xt16   = (h16*)  balloc((size_t)NB*SEQ*DM*2);
  float* xu32   = (float*)balloc((size_t)NB*SEQ*DM*4);
  h16*   xu16   = (h16*)  balloc((size_t)NB*SEQ*DM*2);
  float* ar32   = (float*)balloc((size_t)NB*SEQ*DM*4);      // attn/ffn branch out
  h16*   rows16 = (h16*)  balloc((size_t)NB*SEQ*DM*2);
  float* q32    = (float*)balloc((size_t)NB*SEQ*DM*4);
  h16*   q16    = (h16*)  balloc((size_t)NB*SEQ*DM*2);
  float* kk32   = (float*)balloc((size_t)NB*SEQ*DM*4);
  h16*   kk16   = (h16*)  balloc((size_t)NB*SEQ*DM*2);
  float* v32    = (float*)balloc((size_t)NB*SEQ*DM*4);
  h16*   v16    = (h16*)  balloc((size_t)NB*SEQ*DM*2);
  float* big32  = (float*)balloc((size_t)NB*NH*DECL*DM*4);  // scores / ffn hidden
  h16*   big16  = (h16*)  balloc((size_t)NB*NH*DECL*DM*2);  // probs  / ffn hidden
  float* fo32   = (float*)balloc((size_t)NB*SEQ*DM*4);      // ffn out / conv y
  float* ctx32  = (float*)balloc((size_t)NB*NH*SEQ*HD*4);
  float* mbuf   = (float*)balloc((size_t)NB*NH*SEQ*4);
  int*   top    = (int*)  balloc((size_t)NB*NH*64*4);
  h16*   qred16 = (h16*)  balloc((size_t)NB*NH*64*HD*2);
  float* upd32  = (float*)balloc((size_t)NB*NH*64*HD*4);
  float* meanv  = (float*)balloc((size_t)NB*NH*HD*4);
  h16*   xp16   = (h16*)  balloc((size_t)NB*(SEQ+2)*DM*2);
  float* bnm    = (float*)balloc(DM*4);
  float* bnv    = (float*)balloc(DM*4);
  float* encf32 = (float*)balloc((size_t)NB*512*DM*4);
  h16*   encf16 = (h16*)  balloc((size_t)NB*512*DM*2);

  // ---- f16 weight staging (just-in-time, bump allocated, 256-half slop) ----
  size_t wcur = 0;
  auto walloc = [&](long n) -> h16* {
    h16* d = wbuf + wcur; wcur += (size_t)((n + 255) & ~255L); return d;
  };
  auto cast_w = [&](const float* src, long n) -> const h16* {
    h16* d = walloc(n);
    k_cast<<<g1(n), 256, 0, stream>>>(src, d, n);
    return d;
  };
  auto mk_attn = [&](int base) -> AttnW {   // leaves: bk bo bq bv wk wo wq wv
    AttnW w;
    w.bk = F(base+0); w.bo = F(base+1); w.bq = F(base+2); w.bv = F(base+3);
    w.wk = cast_w(F(base+4), DM*DM); w.wo = cast_w(F(base+5), DM*DM);
    w.wq = cast_w(F(base+6), DM*DM); w.wv = cast_w(F(base+7), DM*DM);
    return w;
  };

  // ---- embedding: circular conv3 (3 accumulated GEMMs, K padded to 32) ----
  auto embed = [&](const float* x, const int* mark, const float* convW,
                   const float* convB, int L) {
    const int CP = 32;   // 14 channels zero-padded to 32
    h16* wr = walloc((long)3 * CP * DM);
    k_repack_convw<<<g1((long)3*CP*DM), 256, 0, stream>>>(convW, wr, 14, CP, (long)3*CP*DM);
    long tpad = (long)NB * (L + 2) * CP;
    k_pad_circ<<<g1(tpad), 256, 0, stream>>>(x, xp16, L, 14, CP, tpad);
    for (int k = 0; k < 3; ++k) {
      GP p{};
      p.A = xp16; p.a0 = (long)k * CP; p.aM = CP; p.aOb = (long)(L+2)*CP;
      p.B = wr;   p.b0 = (long)k * CP * DM; p.bN = 1; p.bK = DM;
      p.C = xe32; p.cM = DM; p.cOb = (long)L * DM;
      p.M = L; p.N = DM; p.K = CP; p.Hdiv = 1; p.alpha = 1.f;
      p.Cold = (k > 0) ? xe32 : nullptr;
      p.bias = (k == 2) ? convB : nullptr;
      gemm_launch(stream, p, NB);
    }
    long tot = (long)NB * L * DM;
    k_embed_add<<<g1(tot), 256, 0, stream>>>(xe32, xe16, mark, L, tot);
  };

  // ---- ProbSparse attention (on xin16, length L) -> aout (B*L,512) f32 ----
  auto prob_attn = [&](const h16* xin16, int L, const AttnW& w, bool mask, bool mix,
                       float* aout) {
    long rows = (long)NB * L; int BH = NB * NH;
    int u = (int)(5.0 * ceil(log((double)L))); if (u > L) u = L;
    gemm_launch(stream, gp_linear(xin16, w.wq, w.bq, q32,  q16,  (int)rows, DM, DM, 0), 1);
    gemm_launch(stream, gp_linear(xin16, w.wk, w.bk, kk32, kk16, (int)rows, DM, DM, 0), 1);
    gemm_launch(stream, gp_linear(xin16, w.wv, w.bv, v32,  v16,  (int)rows, DM, DM, 0), 1);
    k_sample_M<<<dim3(L, BH), 64, 0, stream>>>(q16, kk16, mbuf, L, L, u, NH);
    k_topk<<<BH, 256, 0, stream>>>(mbuf, top, L, u);
    long tg = (long)BH * u * HD;
    k_gather_q<<<g1(tg), 256, 0, stream>>>(q16, top, qred16, L, NH, u, tg);
    { // scores[z][u][L] = Qred @ K^T * 1/sqrt(D)   (B k-contiguous)
      GP p{};
      p.A = qred16; p.aM = HD;          p.aOb = (long)NH*u*HD; p.aOh = (long)u*HD;
      p.B = kk16;   p.bN = DM; p.bK = 1; p.bOb = (long)L*DM;   p.bOh = HD;
      p.C = big32;  p.cM = L;           p.cOb = (long)NH*u*L;  p.cOh = (long)u*L;
      p.M = u; p.N = L; p.K = HD; p.Hdiv = NH; p.alpha = 0.125f;
      gemm_launch(stream, p, BH);
    }
    k_softmax<<<dim3((unsigned)(BH * u)), 256, 0, stream>>>(big32, big16, mask ? top : nullptr, L);
    { // upd[z][u][D] = probs @ V   (B n-contiguous)
      GP p{};
      p.A = big16; p.aM = L;             p.aOb = (long)NH*u*L;  p.aOh = (long)u*L;
      p.B = v16;   p.bN = 1; p.bK = DM;  p.bOb = (long)L*DM;    p.bOh = HD;
      p.C = upd32; p.cM = HD;            p.cOb = (long)NH*u*HD; p.cOh = (long)u*HD;
      p.M = u; p.N = HD; p.K = L; p.Hdiv = NH; p.alpha = 1.f;
      gemm_launch(stream, p, BH);
    }
    if (mask) {
      k_cumsum_ctx<<<BH, 64, 0, stream>>>(v32, ctx32, L, NH);
    } else {
      k_mean_v<<<BH, 64, 0, stream>>>(v32, meanv, L, NH);
      long tf = (long)BH * L * HD;
      k_fill_ctx<<<g1(tf), 256, 0, stream>>>(ctx32, meanv, L, tf);
    }
    k_scatter_ctx<<<g1(tg), 256, 0, stream>>>(ctx32, upd32, top, L, u, tg);
    long tr = rows * DM;
    k_rows_from_ctx<<<g1(tr), 256, 0, stream>>>(ctx32, rows16, L, NH, mix ? 1 : 0, tr);
    gemm_launch(stream, gp_linear(rows16, w.wo, w.bo, aout, nullptr, (int)rows, DM, DM, 0), 1);
  };

  // ---- FFN: f = gelu(x@c1+b1)@c2+b2  -> fo32 ----
  auto ffn = [&](const h16* xin16, int rows, int ibase_c1b) {
    const h16* c1 = cast_w(F(ibase_c1b + 1), (long)DM * DFF);
    const h16* c2 = cast_w(F(ibase_c1b + 3), (long)DFF * DM);
    gemm_launch(stream, gp_linear(xin16, c1, F(ibase_c1b + 0), big32, big16, rows, DFF, DM, 1), 1);
    gemm_launch(stream, gp_linear(big16, c2, F(ibase_c1b + 2), fo32, nullptr, rows, DM, DFF, 0), 1);
  };

  // ============================ ENCODER ============================
  int L = SEQ;
  embed(F(I_XENC), Ii(I_XMARK_ENC), F(I_ENC_EMB_W), F(I_ENC_EMB_B), L);
  for (int li = 0; li < 3; ++li) {
    int eb = (li == 0) ? I_ENC_L0 : (li == 1) ? I_ENC_L1 : I_ENC_L2;
    AttnW aw = mk_attn(eb + 0);
    long rows = (long)NB * L;
    prob_attn(xe16, L, aw, /*mask=*/false, /*mix=*/false, ar32);
    k_add_ln<<<dim3((unsigned)rows), 256, 0, stream>>>(xe32, ar32, F(eb+13), F(eb+12), xt32, xt16);
    ffn(xt16, (int)rows, eb + 8);
    k_add_ln<<<dim3((unsigned)rows), 256, 0, stream>>>(xt32, fo32, F(eb+15), F(eb+14), xe32, xe16);
    if (li < 2) {  // distilling conv layer
      int cb = (li == 0) ? I_ENC_CONV0 : I_ENC_CONV1;   // b, bn_b, bn_g, w
      h16* wc = walloc((long)3 * DM * DM);
      k_repack_convw<<<g1((long)3*DM*DM), 256, 0, stream>>>(F(cb + 3), wc, DM, DM, (long)3*DM*DM);
      long tpad = (long)NB * (L + 2) * DM;
      k_pad_circ<<<g1(tpad), 256, 0, stream>>>(xe32, xp16, L, DM, DM, tpad);
      for (int k = 0; k < 3; ++k) {
        GP p{};
        p.A = xp16; p.a0 = (long)k * DM; p.aM = DM; p.aOb = (long)(L+2)*DM;
        p.B = wc;   p.b0 = (long)k * DM * DM; p.bN = 1; p.bK = DM;
        p.C = fo32; p.cM = DM; p.cOb = (long)L * DM;
        p.M = L; p.N = DM; p.K = DM; p.Hdiv = 1; p.alpha = 1.f;
        p.Cold = (k > 0) ? fo32 : nullptr;
        p.bias = (k == 2) ? F(cb + 0) : nullptr;
        gemm_launch(stream, p, NB);
      }
      k_bn_stats<<<DM, 256, 0, stream>>>(fo32, bnm, bnv, rows);
      long tot = rows * DM;
      k_bn_elu<<<g1(tot), 256, 0, stream>>>(fo32, bnm, bnv, F(cb + 2), F(cb + 1), tot);
      long tpo = (long)NB * (L / 2) * DM;
      k_maxpool<<<g1(tpo), 256, 0, stream>>>(fo32, xe32, xe16, L, tpo);
      L /= 2;
    }
  }
  // final encoder norm -> encf (L == 512)
  k_add_ln<<<dim3((unsigned)(NB * L)), 256, 0, stream>>>(xe32, nullptr, F(I_ENC_NORM_G), F(I_ENC_NORM_B), encf32, encf16);
  const int LS = L;  // 512

  // ============================ DECODER ============================
  const int LD = DECL;
  embed(F(I_XDEC), Ii(I_XMARK_DEC), F(I_DEC_EMB_W), F(I_DEC_EMB_B), LD);
  for (int li = 0; li < 2; ++li) {
    int db = (li == 0) ? I_DEC_L0 : I_DEC_L1;
    long rows = (long)NB * LD;
    // masked self ProbSparse attention (mix=True)
    AttnW sw = mk_attn(db + 18);
    prob_attn(xe16, LD, sw, /*mask=*/true, /*mix=*/true, ar32);
    k_add_ln<<<dim3((unsigned)rows), 256, 0, stream>>>(xe32, ar32, F(db+13), F(db+12), xt32, xt16);
    // full cross attention vs encoder output
    AttnW cw = mk_attn(db + 4);
    gemm_launch(stream, gp_linear(xt16,   cw.wq, cw.bq, q32,  q16,  (int)rows,  DM, DM, 0), 1);
    gemm_launch(stream, gp_linear(encf16, cw.wk, cw.bk, kk32, kk16, NB * LS,    DM, DM, 0), 1);
    gemm_launch(stream, gp_linear(encf16, cw.wv, cw.bv, v32,  v16,  NB * LS,    DM, DM, 0), 1);
    { // scores[z][LD][LS]   (B k-contiguous)
      GP p{};
      p.A = q16;   p.aM = DM;            p.aOb = (long)LD*DM;    p.aOh = HD;
      p.B = kk16;  p.bN = DM; p.bK = 1;  p.bOb = (long)LS*DM;    p.bOh = HD;
      p.C = big32; p.cM = LS;            p.cOb = (long)NH*LD*LS; p.cOh = (long)LD*LS;
      p.M = LD; p.N = LS; p.K = HD; p.Hdiv = NH; p.alpha = 0.125f;
      gemm_launch(stream, p, NB * NH);
    }
    k_softmax<<<dim3((unsigned)(NB * NH * LD)), 256, 0, stream>>>(big32, big16, nullptr, LS);
    { // ctx[z][LD][D] = probs @ V   (B n-contiguous)
      GP p{};
      p.A = big16; p.aM = LS;            p.aOb = (long)NH*LD*LS; p.aOh = (long)LD*LS;
      p.B = v16;   p.bN = 1;  p.bK = DM; p.bOb = (long)LS*DM;    p.bOh = HD;
      p.C = ctx32; p.cM = HD;            p.cOb = (long)NH*LD*HD; p.cOh = (long)LD*HD;
      p.M = LD; p.N = HD; p.K = LS; p.Hdiv = NH; p.alpha = 1.f;
      gemm_launch(stream, p, NB * NH);
    }
    long tr = rows * DM;
    k_rows_from_ctx<<<g1(tr), 256, 0, stream>>>(ctx32, rows16, LD, NH, 0, tr);
    gemm_launch(stream, gp_linear(rows16, cw.wo, cw.bo, ar32, nullptr, (int)rows, DM, DM, 0), 1);
    k_add_ln<<<dim3((unsigned)rows), 256, 0, stream>>>(xt32, ar32, F(db+15), F(db+14), xu32, xu16);
    ffn(xu16, (int)rows, db + 0);
    k_add_ln<<<dim3((unsigned)rows), 256, 0, stream>>>(xu32, fo32, F(db+17), F(db+16), xe32, xe16);
  }
  // final decoder norm
  k_add_ln<<<dim3((unsigned)(NB * LD)), 256, 0, stream>>>(xe32, nullptr, F(I_DEC_NORM_G), F(I_DEC_NORM_B), xt32, xt16);

  // final projection: last PRED_LEN=512 rows per batch -> d_out (B,512,14)
  // weights padded 14 -> 32 columns so K/N staging stays vectorized.
  {
    h16* pw = walloc((long)DM * 32);
    k_pad_w<<<g1((long)DM*32), 256, 0, stream>>>(F(I_PROJ_W), pw, 14, 32, (long)DM*32);
    GP p{};
    p.A = xt16; p.a0 = (long)1024 * DM; p.aM = DM; p.aOb = (long)LD * DM;
    p.B = pw; p.bN = 1; p.bK = 32;
    p.C = (float*)d_out; p.cM = 14; p.cOb = (long)512 * 14;
    p.bias = F(I_PROJ_B);
    p.M = 512; p.N = 14; p.K = DM; p.Hdiv = 1; p.alpha = 1.f;
    gemm_launch(stream, p, NB);
  }
}
